// DeepRetrieval_27539330302432
// MI455X (gfx1250) — matrix operand order repaired and checked by reading, not compiled, not run on hardware
//
#include <hip/hip_runtime.h>
#include <hip/hip_bf16.h>

typedef __attribute__((ext_vector_type(16))) _Float16 v16h;
typedef __attribute__((ext_vector_type(8)))  float    v8f;

#define K_BEAM   5
#define D_EMB    128
#define H_DIM    256
#define N3_OUT   10000
#define B_USERS  512
#define NTILES_OUT (N3_OUT / 16)          // 625
#define NCHUNKS    ((NTILES_OUT + 7) / 8) // 79

union AFrag { v16h v; uint4 q[2]; };

// ---------------------------------------------------------------------------
// Pack fp32 weight [Kd, N] into wave32 WMMA B-fragment panels (f16):
// out[((kt*NT + nt)*32 + lane)*16 + h] = W[kt*32 + (lane>>4)*8 + (h&7) + (h>>3)*16][nt*16 + (lane&15)]
// ---------------------------------------------------------------------------
__global__ void pack_b_kernel(const float* __restrict__ W, _Float16* __restrict__ out,
                              int Kd, int N) {
  int idx = blockIdx.x * blockDim.x + threadIdx.x;
  int total = Kd * N;
  if (idx >= total) return;
  int hh   = idx & 15;
  int lane = (idx >> 4) & 31;
  int t    = idx >> 9;              // kt*NT + nt
  int NT   = N >> 4;
  int nt   = t % NT;
  int kt   = t / NT;
  int col  = nt * 16 + (lane & 15);
  int k    = kt * 32 + (lane >> 4) * 8 + (hh & 7) + (hh >> 3) * 16;
  out[idx] = (_Float16)W[(size_t)k * N + col];
}

// ---------------------------------------------------------------------------
// Beam-input builders (gather + fp32->f16)
// ---------------------------------------------------------------------------
__global__ void build_x1_kernel(const float* __restrict__ user_emb,
                                const int* __restrict__ uidx,
                                _Float16* __restrict__ x1) {
  int idx = blockIdx.x * blockDim.x + threadIdx.x;   // B*128
  if (idx >= B_USERS * D_EMB) return;
  int b = idx >> 7, d = idx & 127;
  x1[idx] = (_Float16)user_emb[(size_t)uidx[b] * D_EMB + d];
}

__global__ void build_x2_kernel(const float* __restrict__ user_emb,
                                const float* __restrict__ node_emb,
                                const int* __restrict__ uidx,
                                const int* __restrict__ i1,
                                _Float16* __restrict__ x2) {
  int idx = blockIdx.x * blockDim.x + threadIdx.x;   // 2560*256
  if (idx >= B_USERS * K_BEAM * 256) return;
  int row = idx >> 8;                                 // b*K + k
  int d   = idx & 255;
  int b   = row / K_BEAM;
  float v = (d < 128) ? user_emb[(size_t)uidx[b] * D_EMB + d]
                      : node_emb[(size_t)i1[row] * D_EMB + (d - 128)];
  x2[idx] = (_Float16)v;
}

__global__ void build_x3_kernel(const float* __restrict__ user_emb,
                                const float* __restrict__ node_emb,
                                const int* __restrict__ uidx,
                                const int* __restrict__ i1,
                                const int* __restrict__ i2,
                                _Float16* __restrict__ x3) {
  int idx = blockIdx.x * blockDim.x + threadIdx.x;   // 12800*384
  if (idx >= B_USERS * K_BEAM * K_BEAM * 384) return;
  int row = idx / 384;                                // (b*K+i)*K + j
  int d   = idx - row * 384;
  int bi  = row / K_BEAM;                             // b*K + i
  int b   = bi / K_BEAM;
  float v;
  if (d < 128)       v = user_emb[(size_t)uidx[b] * D_EMB + d];
  else if (d < 256)  v = node_emb[(size_t)i1[bi] * D_EMB + (d - 128)];
  else               v = node_emb[(size_t)(i2[row] + N3_OUT) * D_EMB + (d - 256)];
  x3[idx] = (_Float16)v;
}

// ---------------------------------------------------------------------------
// Hidden layer: h = BN(relu(x @ W1 + b1)) -> f16 [M, 256]
// One wave computes one 16x16 D tile; grid = (M/16, 4), block = 128 (4 waves).
// ---------------------------------------------------------------------------
template<int KD>
__global__ __launch_bounds__(128) void hidden_mlp_kernel(
    const _Float16* __restrict__ x, const _Float16* __restrict__ pW1,
    const float* __restrict__ b1, const float* __restrict__ gamma,
    const float* __restrict__ beta, const float* __restrict__ mean,
    const float* __restrict__ var, _Float16* __restrict__ h) {
  const int lane  = threadIdx.x & 31;
  const int wave  = __builtin_amdgcn_readfirstlane(threadIdx.x >> 5); // SGPR
  const int mtile = blockIdx.x;
  const int ntile = blockIdx.y * 4 + wave;            // 0..15, scalar
  const int arow  = mtile * 16 + (lane & 15);
  const int kbase = (lane >> 4) * 8;
  const _Float16* xrow = x + (size_t)arow * KD;

  v8f c = {};
  constexpr int KT = KD / 32;
  constexpr int NT = H_DIM / 16;                      // 16
#pragma unroll
  for (int kt = 0; kt < KT; ++kt) {
    AFrag a, b;
    const int ks = kt * 32 + kbase;
    a.q[0] = *(const uint4*)(xrow + ks);
    a.q[1] = *(const uint4*)(xrow + ks + 16);
    const _Float16* bp = pW1 + ((size_t)(kt * NT + ntile) * 32 + lane) * 16;
    b.q[0] = *(const uint4*)(bp);
    b.q[1] = *(const uint4*)(bp + 8);
    c = __builtin_amdgcn_wmma_f32_16x16x32_f16(false, a.v, false, b.v,
                                               (short)0, c, false, false);
  }

  const int   col  = ntile * 16 + (lane & 15);
  const float bb   = b1[col];
  const float g    = gamma[col];
  const float be   = beta[col];
  const float mn   = mean[col];
  const float rstd = rsqrtf(var[col] + 1e-5f);
#pragma unroll
  for (int r = 0; r < 8; ++r) {
    const int m = mtile * 16 + (lane >> 4) * 8 + r;
    float v = fmaxf(c[r] + bb, 0.0f);
    v = (v - mn) * rstd * g + be;
    h[(size_t)m * H_DIM + col] = (_Float16)v;
  }
}

// ---------------------------------------------------------------------------
// Output layer fused with online softmax + top-5:
// logits = h @ W2 + b2 over N=10000, never materialized.
// Block = 256 threads (8 waves) owns a 16-row tile; 79 chunks of 128 cols.
// Scan phase is parallel: 16 rows x 16 column-groups of partial
// online-softmax + partial top-5, merged once per row at the end.
// ---------------------------------------------------------------------------
__global__ __launch_bounds__(256) void out_softmax_topk_kernel(
    const _Float16* __restrict__ h, const _Float16* __restrict__ pW2,
    const float* __restrict__ b2, float* __restrict__ topv,
    int* __restrict__ topi) {
  __shared__ float lbuf[16][129];                     // 16 rows x 128 cols (+pad)
  __shared__ float s_tv[16][16][K_BEAM];
  __shared__ int   s_ti[16][16][K_BEAM];
  __shared__ float s_mx[16][16];
  __shared__ float s_sm[16][16];

  const int tid   = threadIdx.x;
  const int lane  = tid & 31;
  const int wave  = __builtin_amdgcn_readfirstlane(tid >> 5); // SGPR: scalar guard
  const int mtile = blockIdx.x;
  const int arow  = mtile * 16 + (lane & 15);
  const int kbase = (lane >> 4) * 8;

  // Preload full A stripe (16 rows x K=256) into registers: 8 fragments.
  AFrag a[8];
  const _Float16* hrow = h + (size_t)arow * H_DIM;
#pragma unroll
  for (int kt = 0; kt < 8; ++kt) {
    const int ks = kt * 32 + kbase;
    a[kt].q[0] = *(const uint4*)(hrow + ks);
    a[kt].q[1] = *(const uint4*)(hrow + ks + 16);
  }

  // Per-thread partial online-softmax + top-5 (row = tid>>4, colgroup = tid&15).
  const int srow = tid >> 4;
  const int scg  = tid & 15;
  float mx = -3.0e38f, sm = 0.0f;
  float tv[K_BEAM];
  int   ti[K_BEAM];
#pragma unroll
  for (int j = 0; j < K_BEAM; ++j) { tv[j] = -3.0e38f; ti[j] = 0; }

  for (int ch = 0; ch < NCHUNKS; ++ch) {
    const int ntile = ch * 8 + wave;                  // scalar
    if (ntile < NTILES_OUT) {                         // s_cmp/s_cbranch, EXEC intact
      v8f c = {};
#pragma unroll
      for (int kt = 0; kt < 8; ++kt) {
        AFrag b;
        const _Float16* bp =
            pW2 + ((size_t)(kt * NTILES_OUT + ntile) * 32 + lane) * 16;
        b.q[0] = *(const uint4*)(bp);
        b.q[1] = *(const uint4*)(bp + 8);
        c = __builtin_amdgcn_wmma_f32_16x16x32_f16(false, a[kt].v, false, b.v,
                                                   (short)0, c, false, false);
      }
      const int   gcol = ntile * 16 + (lane & 15);
      const float bias = b2[gcol];
      const int   lc   = wave * 16 + (lane & 15);
#pragma unroll
      for (int r = 0; r < 8; ++r)
        lbuf[(lane >> 4) * 8 + r][lc] = c[r] + bias;
      // Prefetch next chunk's B panel (global_prefetch_b8 on gfx1250).
      if (ntile + 8 < NTILES_OUT)
        __builtin_prefetch(pW2 + ((size_t)(ntile + 8) * 32) * 16, 0, 1);
    }
    __syncthreads();
    {
      const int rem   = N3_OUT - ch * 128;
      const int ncols = rem < 128 ? rem : 128;
      for (int cc = scg; cc < ncols; cc += 16) {      // 8 values/thread/chunk
        const float v = lbuf[srow][cc];
        if (v > mx) { sm = sm * __expf(mx - v) + 1.0f; mx = v; }
        else        { sm += __expf(v - mx); }
        if (v > tv[K_BEAM - 1]) {
          int p = K_BEAM - 1;
          while (p > 0 && v > tv[p - 1]) {
            tv[p] = tv[p - 1]; ti[p] = ti[p - 1]; --p;
          }
          tv[p] = v; ti[p] = ch * 128 + cc;
        }
      }
    }
    __syncthreads();
  }

  // Publish partials.
#pragma unroll
  for (int j = 0; j < K_BEAM; ++j) {
    s_tv[srow][scg][j] = tv[j];
    s_ti[srow][scg][j] = ti[j];
  }
  s_mx[srow][scg] = mx;
  s_sm[srow][scg] = sm;
  __syncthreads();

  // Merge 16 partials per row (one thread per row).
  if (tid < 16) {
    const int row = tid;
    float gmx = -3.0e38f;
#pragma unroll
    for (int i = 0; i < 16; ++i) gmx = fmaxf(gmx, s_mx[row][i]);
    float gsm = 0.0f;
#pragma unroll
    for (int i = 0; i < 16; ++i)
      gsm += s_sm[row][i] * __expf(s_mx[row][i] - gmx);

    float mtv[K_BEAM]; int mti[K_BEAM];
#pragma unroll
    for (int j = 0; j < K_BEAM; ++j) { mtv[j] = -3.0e38f; mti[j] = 0; }
    for (int i = 0; i < 16; ++i) {
#pragma unroll
      for (int j = 0; j < K_BEAM; ++j) {
        const float v = s_tv[row][i][j];
        if (v > mtv[K_BEAM - 1]) {
          const int id = s_ti[row][i][j];
          int p = K_BEAM - 1;
          while (p > 0 && v > mtv[p - 1]) {
            mtv[p] = mtv[p - 1]; mti[p] = mti[p - 1]; --p;
          }
          mtv[p] = v; mti[p] = id;
        }
      }
    }
    const int gm = mtile * 16 + row;
    const float inv = 1.0f / gsm;
#pragma unroll
    for (int j = 0; j < K_BEAM; ++j) {
      topv[gm * K_BEAM + j] = __expf(mtv[j] - gmx) * inv;
      topi[gm * K_BEAM + j] = mti[j];
    }
  }
}

// ---------------------------------------------------------------------------
// Combine: top-5 of the 125 path products per user; emit paths + probs.
// ---------------------------------------------------------------------------
__global__ void combine_kernel(const float* __restrict__ p1, const int* __restrict__ i1,
                               const float* __restrict__ p2, const int* __restrict__ i2,
                               const float* __restrict__ p3, const int* __restrict__ i3,
                               float* __restrict__ out) {
  int b = blockIdx.x * blockDim.x + threadIdx.x;
  if (b >= B_USERS) return;
  float tv[K_BEAM]; int tt[K_BEAM];
#pragma unroll
  for (int j = 0; j < K_BEAM; ++j) { tv[j] = -1.0f; tt[j] = 0; }
  for (int i = 0; i < K_BEAM; ++i) {
    const float pa = p1[b * K_BEAM + i];
    for (int j = 0; j < K_BEAM; ++j) {
      const int bij = (b * K_BEAM + i) * K_BEAM + j;
      const float pb = pa * p2[bij];
      for (int k = 0; k < K_BEAM; ++k) {
        const float pp = pb * p3[bij * K_BEAM + k];
        if (pp > tv[K_BEAM - 1]) {
          int p = K_BEAM - 1;
          while (p > 0 && pp > tv[p - 1]) {
            tv[p] = tv[p - 1]; tt[p] = tt[p - 1]; --p;
          }
          tv[p] = pp; tt[p] = (i * K_BEAM + j) * K_BEAM + k;
        }
      }
    }
  }
  for (int q = 0; q < K_BEAM; ++q) {
    const int t = tt[q];
    const int i = t / (K_BEAM * K_BEAM);
    const int j = (t / K_BEAM) % K_BEAM;
    const int k = t % K_BEAM;
    const int n1 = i1[b * K_BEAM + i];
    const int n2 = i2[(b * K_BEAM + i) * K_BEAM + j] + N3_OUT;
    const int n3 = i3[((b * K_BEAM + i) * K_BEAM + j) * K_BEAM + k] + 2 * N3_OUT;
    float* pth = out + (size_t)(b * K_BEAM + q) * 3;
    pth[0] = (float)n1; pth[1] = (float)n2; pth[2] = (float)n3;
    out[B_USERS * K_BEAM * 3 + b * K_BEAM + q] = tv[q];
  }
}

// ---------------------------------------------------------------------------
extern "C" void kernel_launch(void* const* d_in, const int* in_sizes, int n_in,
                              void* d_out, int out_size, void* d_ws, size_t ws_size,
                              hipStream_t stream) {
  (void)in_sizes; (void)n_in; (void)out_size; (void)ws_size;
  const float* user_emb = (const float*)d_in[0];
  const float* node_emb = (const float*)d_in[1];
  const float* W1[3] = { (const float*)d_in[2],  (const float*)d_in[10], (const float*)d_in[18] };
  const float* b1[3] = { (const float*)d_in[3],  (const float*)d_in[11], (const float*)d_in[19] };
  const float* gm[3] = { (const float*)d_in[4],  (const float*)d_in[12], (const float*)d_in[20] };
  const float* bt[3] = { (const float*)d_in[5],  (const float*)d_in[13], (const float*)d_in[21] };
  const float* mn[3] = { (const float*)d_in[6],  (const float*)d_in[14], (const float*)d_in[22] };
  const float* vr[3] = { (const float*)d_in[7],  (const float*)d_in[15], (const float*)d_in[23] };
  const float* W2[3] = { (const float*)d_in[8],  (const float*)d_in[16], (const float*)d_in[24] };
  const float* b2[3] = { (const float*)d_in[9],  (const float*)d_in[17], (const float*)d_in[25] };
  const int*   uidx  = (const int*)d_in[26];
  float* out = (float*)d_out;

  const int Kd[3] = { 128, 256, 384 };
  const int M[3]  = { B_USERS, B_USERS * K_BEAM, B_USERS * K_BEAM * K_BEAM };

  // Workspace bump allocator.
  char* ws = (char*)d_ws;
  size_t off = 0;
  auto alloc = [&](size_t bytes) -> void* {
    void* p = ws + off;
    off = (off + bytes + 255) & ~(size_t)255;
    return p;
  };
  _Float16* x[3];  _Float16* h[3];
  _Float16* pW1[3]; _Float16* pW2[3];
  float* topv[3]; int* topi[3];
  for (int l = 0; l < 3; ++l) {
    x[l]   = (_Float16*)alloc((size_t)M[l] * Kd[l] * sizeof(_Float16));
    h[l]   = (_Float16*)alloc((size_t)M[l] * H_DIM * sizeof(_Float16));
    pW1[l] = (_Float16*)alloc((size_t)Kd[l] * H_DIM * sizeof(_Float16));
    pW2[l] = (_Float16*)alloc((size_t)H_DIM * N3_OUT * sizeof(_Float16));
    topv[l] = (float*)alloc((size_t)M[l] * K_BEAM * sizeof(float));
    topi[l] = (int*)alloc((size_t)M[l] * K_BEAM * sizeof(int));
  }

  // 1) Pack weights into WMMA B-fragment layout (f16).
  for (int l = 0; l < 3; ++l) {
    int t1 = Kd[l] * H_DIM;
    pack_b_kernel<<<(t1 + 255) / 256, 256, 0, stream>>>(W1[l], pW1[l], Kd[l], H_DIM);
    int t2 = H_DIM * N3_OUT;
    pack_b_kernel<<<(t2 + 255) / 256, 256, 0, stream>>>(W2[l], pW2[l], H_DIM, N3_OUT);
  }

  // 2) Level 1
  {
    int t = B_USERS * D_EMB;
    build_x1_kernel<<<(t + 255) / 256, 256, 0, stream>>>(user_emb, uidx, x[0]);
    hidden_mlp_kernel<128><<<dim3(M[0] / 16, 4), 128, 0, stream>>>(
        x[0], pW1[0], b1[0], gm[0], bt[0], mn[0], vr[0], h[0]);
    out_softmax_topk_kernel<<<M[0] / 16, 256, 0, stream>>>(
        h[0], pW2[0], b2[0], topv[0], topi[0]);
  }

  // 3) Level 2
  {
    int t = M[1] * 256;
    build_x2_kernel<<<(t + 255) / 256, 256, 0, stream>>>(
        user_emb, node_emb, uidx, topi[0], x[1]);
    hidden_mlp_kernel<256><<<dim3(M[1] / 16, 4), 128, 0, stream>>>(
        x[1], pW1[1], b1[1], gm[1], bt[1], mn[1], vr[1], h[1]);
    out_softmax_topk_kernel<<<M[1] / 16, 256, 0, stream>>>(
        h[1], pW2[1], b2[1], topv[1], topi[1]);
  }

  // 4) Level 3
  {
    int t = M[2] * 384;
    build_x3_kernel<<<(t + 255) / 256, 256, 0, stream>>>(
        user_emb, node_emb, uidx, topi[0], topi[1], x[2]);
    hidden_mlp_kernel<384><<<dim3(M[2] / 16, 4), 128, 0, stream>>>(
        x[2], pW1[2], b1[2], gm[2], bt[2], mn[2], vr[2], h[2]);
    out_softmax_topk_kernel<<<M[2] / 16, 256, 0, stream>>>(
        h[2], pW2[2], b2[2], topv[2], topi[2]);
  }

  // 5) Combine paths and emit [paths as float | probs].
  combine_kernel<<<(B_USERS + 255) / 256, 256, 0, stream>>>(
      topv[0], topi[0], topv[1], topi[1], topv[2], topi[2], out);
}
